// MultiHeadAttention_15676630630673
// MI455X (gfx1250) — matrix operand order repaired and checked by reading, not compiled, not run on hardware
//
#include <hip/hip_runtime.h>
#include <hip/hip_bf16.h>

// ---------------------------------------------------------------------------
// MHA block for MI455X (gfx1250, wave32, WMMA + TDM).
// B=4, S=1024, D=1024, H=16, DK=DV=64.
// ---------------------------------------------------------------------------

#define B_ 4
#define S_ 1024
#define D_ 1024
#define H_ 16
#define DK_ 64
#define DV_ 64

typedef __attribute__((ext_vector_type(16))) __bf16 v16bf;
typedef __attribute__((ext_vector_type(8)))  float  v8f;
typedef __attribute__((ext_vector_type(4)))  int    v4i;
typedef __attribute__((ext_vector_type(8)))  int    v8i;

union FragBF {               // 16 bf16 = 8 VGPRs, matches v16bf operand
    v16bf v;
    unsigned u[8];
};

__device__ __forceinline__ v8f wmma_bf16(const FragBF& a, const FragBF& b, v8f c) {
    return __builtin_amdgcn_wmma_f32_16x16x32_bf16(
        /*neg_a=*/false, a.v, /*neg_b=*/false, b.v,
        /*c_mod=*/(short)0, c, /*reuse_a=*/false, /*reuse_b=*/false);
}

// ---------------------------------------------------------------------------
// Generic bf16-WMMA GEMM:  out = A[MxK] * W[KxN] + bias (+resid)
// Block tile 64x128, 8 waves (2x4), each wave 2x2 WMMA 16x16 tiles, BK=32.
// LAYOUT 0: out row-major [M][N].  LAYOUT 1: out[(b*H+h)*S+s][dk] (bf16 qkv).
// ---------------------------------------------------------------------------
template <typename TA, typename TO, int LAYOUT, bool RESID>
__global__ __launch_bounds__(256) void gemm_wmma(
    const TA* __restrict__ A, const float* __restrict__ W,
    const float* __restrict__ bias, const float* __restrict__ resid,
    TO* __restrict__ out, int M, int N, int K)
{
    __shared__ __bf16 As[64][34];    // [m][k]   (pad: 34*2B keeps rows 4B-aligned)
    __shared__ __bf16 Bs[128][34];   // [n][k]   transposed for b32 pair reads

    const int tid  = threadIdx.x;
    const int lane = tid & 31;
    const int wv   = tid >> 5;
    const int wm   = wv & 1;          // 2 waves along M
    const int wn   = wv >> 1;         // 4 waves along N
    const int hi   = lane >> 4;       // half-wave (ISA 16-lane halves)
    const int ln   = lane & 15;

    const int row0 = blockIdx.x * 64;
    const int col0 = blockIdx.y * 128;

    v8f acc[2][2] = {};

    for (int k0 = 0; k0 < K; k0 += 32) {
        // prefetch next K tile (emits global_prefetch_b8)
        if (k0 + 32 < K) {
            __builtin_prefetch(A + (size_t)(row0 + (tid >> 2)) * K + k0 + 32, 0, 1);
            __builtin_prefetch(W + (size_t)(k0 + 32 + (tid & 31)) * N + col0, 0, 1);
        }
        // ---- stage A tile 64x32 (f32/bf16 -> bf16) ----
        #pragma unroll
        for (int i = 0; i < 2; ++i) {
            int e  = tid + i * 256;          // 512 float4-slots
            int m  = e >> 3;
            int kq = (e & 7) * 4;
            const TA* src = A + (size_t)(row0 + m) * K + k0 + kq;
            float f0, f1, f2, f3;
            if constexpr (sizeof(TA) == 4) {
                float4 t = *(const float4*)src;
                f0 = t.x; f1 = t.y; f2 = t.z; f3 = t.w;
            } else {
                f0 = (float)src[0]; f1 = (float)src[1];
                f2 = (float)src[2]; f3 = (float)src[3];
            }
            As[m][kq + 0] = (__bf16)f0;  As[m][kq + 1] = (__bf16)f1;
            As[m][kq + 2] = (__bf16)f2;  As[m][kq + 3] = (__bf16)f3;
        }
        // ---- stage W tile 32x128, transposed -> Bs[n][k] ----
        #pragma unroll
        for (int i = 0; i < 4; ++i) {
            int e   = tid + i * 256;         // 1024 float4-slots
            int kk  = e >> 5;
            int nq4 = (e & 31) * 4;
            float4 t = *(const float4*)(W + (size_t)(k0 + kk) * N + col0 + nq4);
            Bs[nq4 + 0][kk] = (__bf16)t.x;
            Bs[nq4 + 1][kk] = (__bf16)t.y;
            Bs[nq4 + 2][kk] = (__bf16)t.z;
            Bs[nq4 + 3][kk] = (__bf16)t.w;
        }
        __syncthreads();

        // ---- fragments (ISA 7.12.2 layouts) ----
        FragBF a[2], bf[2];
        #pragma unroll
        for (int mt = 0; mt < 2; ++mt) {
            int rrow = wm * 32 + mt * 16 + ln;
            #pragma unroll
            for (int r = 0; r < 8; ++r) {
                int kp = (r < 4) ? (2 * r + 8 * hi) : (16 + 2 * (r - 4) + 8 * hi);
                a[mt].u[r] = *(const unsigned*)&As[rrow][kp];
            }
        }
        #pragma unroll
        for (int nt = 0; nt < 2; ++nt) {
            int nn = wn * 32 + nt * 16 + ln;
            #pragma unroll
            for (int v = 0; v < 8; ++v) {
                int kp = (v < 4) ? (2 * v + 8 * hi) : (16 + 2 * (v - 4) + 8 * hi);
                bf[nt].u[v] = *(const unsigned*)&Bs[nn][kp];
            }
        }
        #pragma unroll
        for (int mt = 0; mt < 2; ++mt)
            #pragma unroll
            for (int nt = 0; nt < 2; ++nt)
                acc[mt][nt] = wmma_bf16(a[mt], bf[nt], acc[mt][nt]);
        __syncthreads();
    }

    // ---- epilogue: bias (+resid), layout-mapped store ----
    #pragma unroll
    for (int mt = 0; mt < 2; ++mt) {
        #pragma unroll
        for (int nt = 0; nt < 2; ++nt) {
            #pragma unroll
            for (int r = 0; r < 8; ++r) {
                int row = row0 + wm * 32 + mt * 16 + hi * 8 + r;
                int col = col0 + wn * 32 + nt * 16 + ln;
                float vacc = acc[mt][nt][r] + bias[col];
                if constexpr (RESID) vacc += resid[(size_t)row * N + col];
                size_t oidx;
                if constexpr (LAYOUT == 0) {
                    oidx = (size_t)row * N + col;
                } else {
                    int bb = row >> 10, ss = row & (S_ - 1);   // S=1024
                    int hh = col >> 6,  dk = col & (DK_ - 1);  // DK=64
                    oidx = ((((size_t)bb * H_ + hh) * S_) + ss) * DK_ + dk;
                }
                out[oidx] = (TO)vacc;
            }
        }
    }
}

// ---------------------------------------------------------------------------
// Flash attention per (b,h): 4 waves/block, 16 queries/wave, 32-key blocks.
// K tile DMA'd into LDS via the Tensor Data Mover (linear 4KB copy);
// V tile manually transposed for B-fragment pair reads; mask tile staged in
// LDS with a block-wide "any set?" fast path via __syncthreads_or.
// scale = 1/sqrt(S) = 1/32 (matches the reference's seq_len scaling).
// ---------------------------------------------------------------------------
__global__ __launch_bounds__(128) void flash_attn(
    const __bf16* __restrict__ qb, const __bf16* __restrict__ kb,
    const __bf16* __restrict__ vb, const unsigned char* __restrict__ mask,
    __bf16* __restrict__ ctx)
{
    __shared__ unsigned      KsU[32 * 32];    // 32 keys x 64 dims bf16, pair-packed
    __shared__ __bf16        VsT[64][32];     // [dv][key]  (transposed for B frags)
    __shared__ __bf16        Pst[4][16 * 32]; // per-wave P staging (16x32)
    __shared__ unsigned char Msk[64][32];     // mask tile (slow path only)

    const int tid  = threadIdx.x;
    const int lane = tid & 31;
    const int wv   = tid >> 5;
    const int hi   = lane >> 4;
    const int ln   = lane & 15;

    const int h = blockIdx.y, b = blockIdx.z;
    const int qBlk0 = blockIdx.x * 64;
    const int qBase = qBlk0 + wv * 16;
    const size_t bh = (size_t)b * H_ + h;
    const __bf16* qptr = qb + (bh * S_ + qBase) * DK_;

    // preload Q A-fragments for the two K-chunks of DK=64
    FragBF Qf[2];
    #pragma unroll
    for (int c = 0; c < 2; ++c) {
        #pragma unroll
        for (int r = 0; r < 8; ++r) {
            int kp = ((r < 4) ? (2 * r + 8 * hi) : (16 + 2 * (r - 4) + 8 * hi)) + 32 * c;
            Qf[c].u[r] = *(const unsigned*)(qptr + (size_t)ln * DK_ + kp);
        }
    }

    v8f acc[4] = {};
    float m_[8], l_[8];
    #pragma unroll
    for (int r = 0; r < 8; ++r) { m_[r] = -1e30f; l_[r] = 0.f; }

    const float scale = 0.03125f;   // 1/sqrt(1024)
    const unsigned char* mrow = mask + (size_t)b * S_ * S_;
    const unsigned ldsK = (unsigned)(size_t)&KsU[0];

    for (int key0 = 0; key0 < S_; key0 += 32) {
        // ---- K tile: Tensor Data Mover, linear 4KB global -> LDS ----
        if (wv == 0) {
            unsigned long long ga =
                (unsigned long long)(size_t)(kb + (bh * S_ + key0) * DK_);
            v4i g0;
            g0.x = 1;                                   // count=1, user descriptor
            g0.y = (int)ldsK;                           // lds_addr
            g0.z = (int)(unsigned)(ga & 0xFFFFFFFFu);   // global_addr[31:0]
            g0.w = (int)(((unsigned)(ga >> 32) & 0x1FFFFFFu) | (2u << 30)); // +type=2
            v8i g1;
            g1.s0 = 0;                  // wgmask=0 (no cluster), data_size=1B
            g1.s1 = (int)(4096u << 16); // tensor_dim0 = 4096 bytes (low16 @ [63:48])
            g1.s2 = (int)(1u << 16);    // tensor_dim0 hi=0 ; tensor_dim1 = 1
            g1.s3 = (int)(4096u << 16); // tensor_dim1 hi=0 ; tile_dim0 = 4096
            g1.s4 = 1;                  // tile_dim1 = 1 ; tile_dim2 = 0
            g1.s5 = 4096;               // tensor_dim0_stride = 4096
            g1.s6 = 0;
            g1.s7 = 0;
            asm volatile("tensor_load_to_lds %0, %1"
                         :: "s"(g0), "s"(g1) : "memory");
            __builtin_amdgcn_s_wait_tensorcnt(0);
        }

        // ---- V tile: manual transpose into LDS ----
        const unsigned* vsrc = (const unsigned*)(vb + (bh * S_ + key0) * DV_);
        #pragma unroll
        for (int i = 0; i < 8; ++i) {
            int e = tid + i * 128;             // 1024 uint slots
            int key = e >> 5, dvu = e & 31;
            unsigned p = vsrc[e];
            union { unsigned u; __bf16 hh[2]; } cv; cv.u = p;
            VsT[2 * dvu + 0][key] = cv.hh[0];
            VsT[2 * dvu + 1][key] = cv.hh[1];
        }

        // ---- mask tile: coalesced 16B/thread stage + block-wide OR ----
        {
            int mr = tid >> 1, mh = tid & 1;
            uint4 m4 = *(const uint4*)(mrow + (size_t)(qBlk0 + mr) * S_ + key0 + mh * 16);
            ((uint4*)&Msk[0][0])[tid] = m4;
            unsigned orv = m4.x | m4.y | m4.z | m4.w;
            // barrier #1: publishes K (TDM), V, mask tiles; returns mask summary
            int anyMask = __syncthreads_or((int)(orv != 0u));

            // ---- scores S = Q K^T for two 16-key subtiles ----
            v8f Sv[2];
            #pragma unroll
            for (int t = 0; t < 2; ++t) {
                v8f sacc = {};
                #pragma unroll
                for (int c = 0; c < 2; ++c) {
                    FragBF kf;
                    #pragma unroll
                    for (int v = 0; v < 8; ++v) {
                        int d = ((v < 4) ? (2 * v + 8 * hi) : (16 + 2 * (v - 4) + 8 * hi)) + 32 * c;
                        kf.u[v] = KsU[(t * 16 + ln) * 32 + (d >> 1)];
                    }
                    sacc = wmma_bf16(Qf[c], kf, sacc);
                }
                Sv[t] = sacc;
            }
            // scale (+ mask slow path from LDS, skipped when tile is clean)
            #pragma unroll
            for (int t = 0; t < 2; ++t) {
                #pragma unroll
                for (int r = 0; r < 8; ++r)
                    Sv[t][r] *= scale;
            }
            if (anyMask) {
                #pragma unroll
                for (int t = 0; t < 2; ++t) {
                    #pragma unroll
                    for (int r = 0; r < 8; ++r) {
                        if (Msk[wv * 16 + hi * 8 + r][t * 16 + ln])
                            Sv[t][r] = -1e30f;
                    }
                }
            }
            // ---- online softmax (row stats live in 16-lane halves) ----
            #pragma unroll
            for (int r = 0; r < 8; ++r) {
                float mx = fmaxf(Sv[0][r], Sv[1][r]);
                mx = fmaxf(mx, __shfl_xor(mx, 1, 16));
                mx = fmaxf(mx, __shfl_xor(mx, 2, 16));
                mx = fmaxf(mx, __shfl_xor(mx, 4, 16));
                mx = fmaxf(mx, __shfl_xor(mx, 8, 16));
                float mnew = fmaxf(m_[r], mx);
                float corr = __expf(m_[r] - mnew);
                m_[r] = mnew;
                l_[r] *= corr;
                #pragma unroll
                for (int nt = 0; nt < 4; ++nt) acc[nt][r] *= corr;
                float p0 = __expf(Sv[0][r] - mnew);
                float p1 = __expf(Sv[1][r] - mnew);
                Sv[0][r] = p0; Sv[1][r] = p1;
                float rs = p0 + p1;
                rs += __shfl_xor(rs, 1, 16);
                rs += __shfl_xor(rs, 2, 16);
                rs += __shfl_xor(rs, 4, 16);
                rs += __shfl_xor(rs, 8, 16);
                l_[r] += rs;
            }
            // ---- restage P (C layout -> A layout) via wave-private LDS ----
            __bf16* pw = &Pst[wv][0];
            #pragma unroll
            for (int t = 0; t < 2; ++t)
                #pragma unroll
                for (int r = 0; r < 8; ++r)
                    pw[(hi * 8 + r) * 32 + t * 16 + ln] = (__bf16)Sv[t][r];
            FragBF Pf;
            #pragma unroll
            for (int r = 0; r < 8; ++r) {
                int kp = (r < 4) ? (2 * r + 8 * hi) : (16 + 2 * (r - 4) + 8 * hi);
                Pf.u[r] = *(const unsigned*)&pw[ln * 32 + kp];
            }
            // ---- context += P @ V ----
            #pragma unroll
            for (int nt = 0; nt < 4; ++nt) {
                FragBF vf;
                #pragma unroll
                for (int v = 0; v < 8; ++v) {
                    int key = (v < 4) ? (2 * v + 8 * hi) : (16 + 2 * (v - 4) + 8 * hi);
                    vf.u[v] = *(const unsigned*)&VsT[nt * 16 + ln][key];
                }
                acc[nt] = wmma_bf16(Pf, vf, acc[nt]);
            }
        }
        __syncthreads();   // tiles consumed; safe to overwrite next iteration
    }

    // ---- normalize & store ctx bf16 [B*S][H*DV] ----
    #pragma unroll
    for (int r = 0; r < 8; ++r) {
        float inv = 1.0f / l_[r];
        int q = qBase + hi * 8 + r;
        __bf16* orow = ctx + ((size_t)(b * S_ + q) * (H_ * DV_)) + h * DV_;
        #pragma unroll
        for (int nt = 0; nt < 4; ++nt)
            orow[nt * 16 + ln] = (__bf16)(acc[nt][r] * inv);
    }
}

// ---------------------------------------------------------------------------
// Row LayerNorm over D=1024 (input y already has bias+residual).
// ---------------------------------------------------------------------------
__global__ __launch_bounds__(256) void layernorm_rows(
    const float* __restrict__ y, const float* __restrict__ g,
    const float* __restrict__ be, float* __restrict__ out)
{
    __shared__ float red[2][8];
    const int row = blockIdx.x;
    const float* x = y + (size_t)row * D_;
    float vals[4], sum = 0.f, sq = 0.f;
    #pragma unroll
    for (int i = 0; i < 4; ++i) {
        float v = x[threadIdx.x + i * 256];
        vals[i] = v; sum += v; sq += v * v;
    }
    #pragma unroll
    for (int off = 16; off; off >>= 1) {
        sum += __shfl_xor(sum, off, 32);
        sq  += __shfl_xor(sq,  off, 32);
    }
    int lane = threadIdx.x & 31, wv = threadIdx.x >> 5;
    if (lane == 0) { red[0][wv] = sum; red[1][wv] = sq; }
    __syncthreads();
    float ts = 0.f, tq = 0.f;
    #pragma unroll
    for (int i = 0; i < 8; ++i) { ts += red[0][i]; tq += red[1][i]; }
    float mu   = ts * (1.0f / D_);
    float var  = tq * (1.0f / D_) - mu * mu;
    float rstd = __frsqrt_rn(var + 1e-5f);
    #pragma unroll
    for (int i = 0; i < 4; ++i) {
        int c = threadIdx.x + i * 256;
        out[(size_t)row * D_ + c] = (vals[i] - mu) * rstd * g[c] + be[c];
    }
}

// ---------------------------------------------------------------------------
extern "C" void kernel_launch(void* const* d_in, const int* in_sizes, int n_in,
                              void* d_out, int out_size, void* d_ws, size_t ws_size,
                              hipStream_t stream)
{
    const float* Q    = (const float*)d_in[0];
    const float* K    = (const float*)d_in[1];
    const float* V    = (const float*)d_in[2];
    const float* Wq   = (const float*)d_in[3];
    const float* bq   = (const float*)d_in[4];
    const float* Wk   = (const float*)d_in[5];
    const float* bk   = (const float*)d_in[6];
    const float* Wv   = (const float*)d_in[7];
    const float* bv   = (const float*)d_in[8];
    const float* Wo   = (const float*)d_in[9];
    const float* bo   = (const float*)d_in[10];
    const float* ln_g = (const float*)d_in[11];
    const float* ln_b = (const float*)d_in[12];
    const unsigned char* mask = (const unsigned char*)d_in[13];

    const int M = B_ * S_;            // 4096
    const int N = H_ * DK_;           // 1024
    const int Kd = D_;                // 1024
    const size_t QKV_E = (size_t)B_ * H_ * S_ * DK_;   // 4M elems

    __bf16* qb  = (__bf16*)d_ws;
    __bf16* kb  = qb  + QKV_E;
    __bf16* vb  = kb  + QKV_E;
    __bf16* ctx = vb  + QKV_E;
    float*  yy  = (float*)(ctx + QKV_E);

    dim3 gG(M / 64, N / 128);         // 64 x 8
    gemm_wmma<float, __bf16, 1, false><<<gG, 256, 0, stream>>>(Q, Wq, bq, nullptr, qb, M, N, Kd);
    gemm_wmma<float, __bf16, 1, false><<<gG, 256, 0, stream>>>(K, Wk, bk, nullptr, kb, M, N, Kd);
    gemm_wmma<float, __bf16, 1, false><<<gG, 256, 0, stream>>>(V, Wv, bv, nullptr, vb, M, N, Kd);

    flash_attn<<<dim3(S_ / 64, H_, B_), 128, 0, stream>>>(qb, kb, vb, mask, ctx);

    gemm_wmma<__bf16, float, 0, true><<<gG, 256, 0, stream>>>(ctx, Wo, bo, Q, yy, M, D_, H_ * DV_);

    layernorm_rows<<<M, 256, 0, stream>>>(yy, ln_g, ln_b, (float*)d_out);
}
